// CombinedSparseAdaptiveAttention_49890340110533
// MI455X (gfx1250) — matrix operand order repaired and checked by reading, not compiled, not run on hardware
//
#include <hip/hip_runtime.h>
#include <hip/hip_bf16.h>
#include <math.h>

// ---------------------------------------------------------------------------
// CombinedSparseAdaptiveAttention for MI455X (gfx1250, wave32, WMMA)
// B=4, S=4096, D=1024, HEAD=16, dh=64, k_top = 1024 (SPARSITY=0.25, span>k)
// ---------------------------------------------------------------------------

typedef __bf16 bf16_t;
typedef __attribute__((ext_vector_type(16))) __bf16 v16bf;
typedef __attribute__((ext_vector_type(8)))  __bf16 v8bf;
typedef __attribute__((ext_vector_type(8)))  float  v8f;

#define B_DIM   4
#define S_DIM   4096
#define D_MODEL 1024
#define N_TOK   1024   // tokens kept by top-k
#define N_HEAD  16
#define D_HEAD  64

static __device__ __forceinline__ v16bf cat8(v8bf lo, v8bf hi) {
  return __builtin_shufflevector(lo, hi, 0,1,2,3,4,5,6,7,8,9,10,11,12,13,14,15);
}

// ---------------------------------------------------------------------------
// 1) L2 norm of each query row (one wave32 per row)
// ---------------------------------------------------------------------------
__global__ void row_norm_kernel(const float* __restrict__ x, float* __restrict__ norms) {
  int wid  = threadIdx.x >> 5;
  int lane = threadIdx.x & 31;
  int row  = blockIdx.x * (blockDim.x >> 5) + wid;
  if (row >= B_DIM * S_DIM) return;
  const float* p = x + (size_t)row * D_MODEL;
  float s = 0.f;
  for (int i = lane; i < D_MODEL; i += 32) { float v = p[i]; s += v * v; }
  #pragma unroll
  for (int m = 16; m >= 1; m >>= 1) s += __shfl_xor(s, m, 32);
  if (lane == 0) norms[row] = sqrtf(s);
}

// ---------------------------------------------------------------------------
// 2) Exact top-1024 of 4096 per batch: in-LDS bitonic sort (desc, idx-asc ties)
// ---------------------------------------------------------------------------
__global__ void topk_kernel(const float* __restrict__ norms, int* __restrict__ idx_out) {
  __shared__ float key[S_DIM];
  __shared__ int   val[S_DIM];
  const int b = blockIdx.x, tid = threadIdx.x;
  for (int i = tid; i < S_DIM; i += blockDim.x) { key[i] = norms[b * S_DIM + i]; val[i] = i; }
  __syncthreads();
  for (int k = 2; k <= S_DIM; k <<= 1) {
    for (int j = k >> 1; j > 0; j >>= 1) {
      for (int i = tid; i < S_DIM; i += blockDim.x) {
        int ixj = i ^ j;
        if (ixj > i) {
          float ki = key[i], kj = key[ixj];
          int   vi = val[i], vj = val[ixj];
          bool igreater = (ki > kj) || (ki == kj && vi < vj);
          bool do_swap  = ((i & k) == 0) ? (!igreater) : igreater;
          if (do_swap) { key[i] = kj; key[ixj] = ki; val[i] = vj; val[ixj] = vi; }
        }
      }
      __syncthreads();
    }
  }
  for (int i = tid; i < N_TOK; i += blockDim.x) idx_out[b * N_TOK + i] = val[i];
}

// ---------------------------------------------------------------------------
// 3) Gather selected rows and cast fp32 -> bf16
// ---------------------------------------------------------------------------
__global__ void gather_cast_kernel(const float* __restrict__ src, const int* __restrict__ idx,
                                   bf16_t* __restrict__ dst) {
  int row = blockIdx.x;                 // 0 .. B*N_TOK-1
  int b = row >> 10, n = row & 1023;
  int srow = idx[b * N_TOK + n];
  const float* s = src + ((size_t)b * S_DIM + srow) * D_MODEL;
  bf16_t*      d = dst + (size_t)row * D_MODEL;
  for (int i = threadIdx.x; i < D_MODEL; i += blockDim.x) d[i] = (bf16_t)s[i];
}

__global__ void cast_bf16_kernel(const float* __restrict__ src, bf16_t* __restrict__ dst, int n) {
  int i = blockIdx.x * blockDim.x + threadIdx.x;
  if (i < n) dst[i] = (bf16_t)src[i];
}

__global__ void zero_kernel(float* __restrict__ p, int n) {
  int i = blockIdx.x * blockDim.x + threadIdx.x;
  if (i < n) p[i] = 0.f;
}

// ---------------------------------------------------------------------------
// 4/6) WMMA bf16 GEMM, NT form: C[m,n] = sum_k A[m,k]*Bw[n,k] + bias[n]
// One wave computes a 16x64 C tile (4 accumulators): each A fragment is
// reused across 4 B fragments -> 4 WMMAs per 10 b128 loads per k-step.
// Optional outputs: fp32 (Cf), bf16 (Cbf), and per-batch transposed bf16 (Ct):
//   Ct[(m/1024)][n][m%1024]  (V transposed per batch for the PV GEMM)
// ---------------------------------------------------------------------------
__global__ void gemm_bf16_wmma(const bf16_t* __restrict__ A, const bf16_t* __restrict__ Bw,
                               const float* __restrict__ bias,
                               float* __restrict__ Cf, bf16_t* __restrict__ Cbf,
                               bf16_t* __restrict__ Ct,
                               int M, int N, int K) {
  const int wid  = threadIdx.x >> 5;
  const int lane = threadIdx.x & 31;
  const int half = lane >> 4, l16 = lane & 15;
  const int n64tiles = N >> 6;
  int tile = blockIdx.x * 4 + wid;
  int mt = tile / n64tiles, nt64 = tile % n64tiles;
  if (mt >= (M >> 4)) return;

  const bf16_t* arow = A + (size_t)(mt * 16 + l16) * K;
  const bf16_t* brow0 = Bw + (size_t)(nt64 * 64 +  0 + l16) * K;
  const bf16_t* brow1 = Bw + (size_t)(nt64 * 64 + 16 + l16) * K;
  const bf16_t* brow2 = Bw + (size_t)(nt64 * 64 + 32 + l16) * K;
  const bf16_t* brow3 = Bw + (size_t)(nt64 * 64 + 48 + l16) * K;

  v8f acc0 = {0.f,0.f,0.f,0.f,0.f,0.f,0.f,0.f};
  v8f acc1 = acc0, acc2 = acc0, acc3 = acc0;

  for (int k = 0; k < K; k += 32) {
    __builtin_prefetch(arow + k + 256, 0, 1);
    // A fragment (16x32): lane half selects contiguous 16 K values
    v8bf a0 = *(const v8bf*)(arow + k + half * 16);
    v8bf a1 = *(const v8bf*)(arow + k + half * 16 + 8);
    v16bf af = cat8(a0, a1);
    // B fragments (32x16): lane half selects contiguous 16 K values
    const int kb = k + half * 16;
    v16bf bf0 = cat8(*(const v8bf*)(brow0 + kb), *(const v8bf*)(brow0 + kb + 8));
    v16bf bf1 = cat8(*(const v8bf*)(brow1 + kb), *(const v8bf*)(brow1 + kb + 8));
    v16bf bf2 = cat8(*(const v8bf*)(brow2 + kb), *(const v8bf*)(brow2 + kb + 8));
    v16bf bf3 = cat8(*(const v8bf*)(brow3 + kb), *(const v8bf*)(brow3 + kb + 8));
    acc0 = __builtin_amdgcn_wmma_f32_16x16x32_bf16(false, af, false, bf0, (short)0, acc0, false, false);
    acc1 = __builtin_amdgcn_wmma_f32_16x16x32_bf16(false, af, false, bf1, (short)0, acc1, false, false);
    acc2 = __builtin_amdgcn_wmma_f32_16x16x32_bf16(false, af, false, bf2, (short)0, acc2, false, false);
    acc3 = __builtin_amdgcn_wmma_f32_16x16x32_bf16(false, af, false, bf3, (short)0, acc3, false, false);
  }

  v8f accs[4] = {acc0, acc1, acc2, acc3};
  #pragma unroll
  for (int j = 0; j < 4; ++j) {
    const int ncol = nt64 * 64 + j * 16 + l16;
    const float bv = bias ? bias[ncol] : 0.f;
    #pragma unroll
    for (int r = 0; r < 8; ++r) {
      int m = mt * 16 + r + half * 8;          // C layout: M = r + 8*(lane>=16)
      float v = accs[j][r] + bv;
      if (Cf)  Cf[(size_t)m * N + ncol] = v;
      if (Cbf) Cbf[(size_t)m * N + ncol] = (bf16_t)v;
      if (Ct)  Ct[(size_t)(m >> 10) * ((size_t)N * N_TOK) + (size_t)ncol * N_TOK + (m & 1023)] = (bf16_t)v;
    }
  }
}

// ---------------------------------------------------------------------------
// 5) Attention: one block per (b, h, 16-row q tile); 4 waves (128 threads).
//    scores (WMMA, hoisted Q fragments) -> fp32 LDS -> softmax ->
//    attn-mean atomicAdd -> PV (WMMA)
// Dynamic LDS: 16*1024 fp32 scores + reduction scratch.
// ---------------------------------------------------------------------------
__global__ void attn_kernel(const bf16_t* __restrict__ Q, const bf16_t* __restrict__ Kb,
                            const bf16_t* __restrict__ Vt,
                            bf16_t* __restrict__ Oa, float* __restrict__ attn_mean) {
  extern __shared__ char smem_raw[];
  float* sc   = (float*)smem_raw;                                   // [16][1024]
  float* red  = (float*)(smem_raw + 16 * 1024 * sizeof(float));     // [16][8]
  float* rmax = red + 128;                                          // [16]
  float* rsum = rmax + 16;                                          // [16]

  const int qt = blockIdx.x, h = blockIdx.y, b = blockIdx.z;
  const int tid = threadIdx.x, wid = tid >> 5, lane = tid & 31;
  const int half = lane >> 4, l16 = lane & 15;
  const int q0 = qt * 16;

  const bf16_t* Qb    = Q  + ((size_t)b * N_TOK + q0) * D_MODEL + h * D_HEAD;
  const bf16_t* Kbase = Kb + (size_t)b * N_TOK * D_MODEL + h * D_HEAD;
  const bf16_t* Vbase = Vt + ((size_t)b * D_MODEL + h * D_HEAD) * N_TOK;

  // ---- scores = (Q · K^T) * 1/sqrt(dh); M=16 q rows, N=1024 keys, K=64 ----
  // Q fragments are loop-invariant (dh = 64 -> exactly 2 k-steps): hoist them.
  const bf16_t* arow = Qb + (size_t)l16 * D_MODEL;
  const v16bf afk0 = cat8(*(const v8bf*)(arow + half * 16),
                          *(const v8bf*)(arow + 8 + half * 16));
  const v16bf afk1 = cat8(*(const v8bf*)(arow + 32 + half * 16),
                          *(const v8bf*)(arow + 40 + half * 16));
  for (int nt = wid; nt < N_TOK / 16; nt += 4) {
    const bf16_t* brow = Kbase + (size_t)(nt * 16 + l16) * D_MODEL;
    v8f acc = {0.f,0.f,0.f,0.f,0.f,0.f,0.f,0.f};
    const int kb = half * 16;
    v16bf bf0 = cat8(*(const v8bf*)(brow + kb),      *(const v8bf*)(brow + kb + 8));
    v16bf bf1 = cat8(*(const v8bf*)(brow + 32 + kb), *(const v8bf*)(brow + 32 + kb + 8));
    acc = __builtin_amdgcn_wmma_f32_16x16x32_bf16(false, afk0, false, bf0, (short)0, acc, false, false);
    acc = __builtin_amdgcn_wmma_f32_16x16x32_bf16(false, afk1, false, bf1, (short)0, acc, false, false);
    #pragma unroll
    for (int r = 0; r < 8; ++r)
      sc[(r + half * 8) * N_TOK + nt * 16 + l16] = acc[r] * 0.125f;  // 1/sqrt(64)
  }
  __syncthreads();

  // ---- softmax over 1024 keys; 8 threads per row ----
  const int row = tid >> 3, sub = tid & 7;
  float* srow = sc + row * N_TOK;
  float mx = -3.4e38f;
  for (int c = sub * 128; c < sub * 128 + 128; ++c) mx = fmaxf(mx, srow[c]);
  red[row * 8 + sub] = mx;
  __syncthreads();
  if (sub == 0) {
    float m = red[row * 8];
    #pragma unroll
    for (int i = 1; i < 8; ++i) m = fmaxf(m, red[row * 8 + i]);
    rmax[row] = m;
  }
  __syncthreads();
  float m = rmax[row], s = 0.f;
  for (int c = sub * 128; c < sub * 128 + 128; ++c) {
    float e = __expf(srow[c] - m);
    srow[c] = e; s += e;
  }
  red[row * 8 + sub] = s;
  __syncthreads();
  if (sub == 0) {
    float t = 0.f;
    #pragma unroll
    for (int i = 0; i < 8; ++i) t += red[row * 8 + i];
    rsum[row] = t;
  }
  __syncthreads();
  const float inv = 1.f / rsum[row];
  float* am = attn_mean + ((size_t)b * N_TOK + q0 + row) * N_TOK;
  for (int c = sub * 128; c < sub * 128 + 128; ++c) {
    float p = srow[c] * inv;
    srow[c] = p;
    atomicAdd(&am[c], p * (1.f / (float)N_HEAD));
  }
  __syncthreads();

  // ---- out tile = P · V : M=16, N=64 (wave wid owns 16 dh cols), K=1024 ----
  {
    const bf16_t* vrow = Vbase + (size_t)(wid * 16 + l16) * N_TOK;  // Vt row = fixed dh col
    const float*  ar   = sc + (size_t)l16 * N_TOK;
    v8f acc = {0.f,0.f,0.f,0.f,0.f,0.f,0.f,0.f};
    for (int k = 0; k < N_TOK; k += 32) {
      v8bf a0, a1;
      const int ka = k + half * 16;
      #pragma unroll
      for (int j = 0; j < 8; ++j) {
        a0[j] = (bf16_t)ar[ka + j];
        a1[j] = (bf16_t)ar[ka + 8 + j];
      }
      v16bf bfm = cat8(*(const v8bf*)(vrow + k + half * 16),
                       *(const v8bf*)(vrow + k + half * 16 + 8));
      acc = __builtin_amdgcn_wmma_f32_16x16x32_bf16(false, cat8(a0, a1),
                                                    false, bfm,
                                                    (short)0, acc, false, false);
    }
    bf16_t* ob = Oa + ((size_t)b * N_TOK + q0) * D_MODEL + h * D_HEAD + wid * 16 + l16;
    #pragma unroll
    for (int r = 0; r < 8; ++r)
      ob[(size_t)(r + half * 8) * D_MODEL] = (bf16_t)acc[r];
  }
}

// ---------------------------------------------------------------------------
// Host-side launch
// ---------------------------------------------------------------------------
extern "C" void kernel_launch(void* const* d_in, const int* in_sizes, int n_in,
                              void* d_out, int out_size, void* d_ws, size_t ws_size,
                              hipStream_t stream) {
  (void)in_sizes; (void)n_in; (void)out_size; (void)ws_size;
  const float* q_in   = (const float*)d_in[0];
  const float* k_in   = (const float*)d_in[1];
  const float* v_in   = (const float*)d_in[2];
  const float* inpw   = (const float*)d_in[3];   // [3072,1024]
  const float* inpb   = (const float*)d_in[4];   // [3072]
  const float* outw   = (const float*)d_in[5];   // [1024,1024]
  const float* outb   = (const float*)d_in[6];   // [1024]
  float* out = (float*)d_out;                    // [4*1024*1024 out | 4*1024*1024 attn_mean]

  // workspace carve-up
  char* ws = (char*)d_ws;
  size_t cur = 0;
  auto alloc = [&](size_t bytes) { char* p = ws + cur; cur += (bytes + 255) & ~(size_t)255; return p; };
  float*  norms = (float*)alloc((size_t)B_DIM * S_DIM * 4);
  int*    idx   = (int*)  alloc((size_t)B_DIM * N_TOK * 4);
  bf16_t* Xq    = (bf16_t*)alloc((size_t)B_DIM * N_TOK * D_MODEL * 2);
  bf16_t* Xk    = (bf16_t*)alloc((size_t)B_DIM * N_TOK * D_MODEL * 2);
  bf16_t* Xv    = (bf16_t*)alloc((size_t)B_DIM * N_TOK * D_MODEL * 2);
  bf16_t* Wbf   = (bf16_t*)alloc((size_t)3 * D_MODEL * D_MODEL * 2);  // wq|wk|wv
  bf16_t* OWbf  = (bf16_t*)alloc((size_t)D_MODEL * D_MODEL * 2);
  bf16_t* Qbf   = (bf16_t*)alloc((size_t)B_DIM * N_TOK * D_MODEL * 2);
  bf16_t* Kbf   = (bf16_t*)alloc((size_t)B_DIM * N_TOK * D_MODEL * 2);
  bf16_t* Vtbf  = (bf16_t*)alloc((size_t)B_DIM * D_MODEL * N_TOK * 2); // [b][dmodel][tok]
  bf16_t* Oa    = (bf16_t*)alloc((size_t)B_DIM * N_TOK * D_MODEL * 2);

  // 1) query row norms
  row_norm_kernel<<<(B_DIM * S_DIM) / 8, 256, 0, stream>>>(q_in, norms);
  // 2) top-k indices per batch
  topk_kernel<<<B_DIM, 1024, 0, stream>>>(norms, idx);
  // 3) gather + cast
  gather_cast_kernel<<<B_DIM * N_TOK, 256, 0, stream>>>(q_in, idx, Xq);
  gather_cast_kernel<<<B_DIM * N_TOK, 256, 0, stream>>>(k_in, idx, Xk);
  gather_cast_kernel<<<B_DIM * N_TOK, 256, 0, stream>>>(v_in, idx, Xv);
  cast_bf16_kernel<<<(3 * D_MODEL * D_MODEL) / 256, 256, 0, stream>>>(inpw, Wbf, 3 * D_MODEL * D_MODEL);
  cast_bf16_kernel<<<(D_MODEL * D_MODEL) / 256, 256, 0, stream>>>(outw, OWbf, D_MODEL * D_MODEL);

  // 4) projections: M=4096, N=1024, K=1024 ; wave = 16x64 tile, 4 waves/block
  const int M = B_DIM * N_TOK, N = D_MODEL, K = D_MODEL;
  const int gblocks = (M / 16) * (N / 64) / 4;
  gemm_bf16_wmma<<<gblocks, 128, 0, stream>>>(Xq, Wbf,                       inpb,            nullptr, Qbf, nullptr, M, N, K);
  gemm_bf16_wmma<<<gblocks, 128, 0, stream>>>(Xk, Wbf + (size_t)N * K,       inpb + D_MODEL,  nullptr, Kbf, nullptr, M, N, K);
  gemm_bf16_wmma<<<gblocks, 128, 0, stream>>>(Xv, Wbf + (size_t)2 * N * K,   inpb + 2*D_MODEL,nullptr, nullptr, Vtbf, M, N, K);

  // 5) attention (+ attn-mean accumulation into second half of d_out)
  float* attn_mean = out + (size_t)B_DIM * N_TOK * D_MODEL;
  zero_kernel<<<(B_DIM * N_TOK * N_TOK) / 256, 256, 0, stream>>>(attn_mean, B_DIM * N_TOK * N_TOK);
  const size_t attn_lds = 16 * 1024 * sizeof(float) + (128 + 32) * sizeof(float);
  attn_kernel<<<dim3(N_TOK / 16, N_HEAD, B_DIM), 128, attn_lds, stream>>>(Qbf, Kbf, Vtbf, Oa, attn_mean);

  // 6) output projection -> fp32 into first half of d_out
  gemm_bf16_wmma<<<gblocks, 128, 0, stream>>>(Oa, OWbf, outb, out, nullptr, nullptr, M, N, K);
}
